// Gemma4MoE_75849122447854
// MI455X (gfx1250) — matrix operand order repaired and checked by reading, not compiled, not run on hardware
//
#include <hip/hip_runtime.h>
#include <hip/hip_bf16.h>
#include <math.h>

// Problem constants (from reference)
#define TT   4096          // B*S tokens
#define DD   1024          // EMB
#define FF   2048          // SHARED_MLP
#define MM   1024          // MOE_MLP
#define NE   8             // experts
#define TOPK 2
#define EPSV 1e-6f

typedef __bf16 bf16x16 __attribute__((ext_vector_type(16)));
typedef float  f32x8   __attribute__((ext_vector_type(8)));

union FragBF { uint4 q[2]; bf16x16 v; };

// --- CDNA5 async global->LDS copy (ASYNCcnt), guarded so the file compiles
// --- on toolchains without the builtin (falls back to VGPR round-trip).
#if defined(__has_builtin)
#  if __has_builtin(__builtin_amdgcn_global_load_async_to_lds_b128)
#    define HAVE_ASYNC_LDS 1
#  endif
#endif
#ifndef HAVE_ASYNC_LDS
#  define HAVE_ASYNC_LDS 0
#endif

#if HAVE_ASYNC_LDS
typedef int v4i_vs __attribute__((vector_size(16)));
typedef __attribute__((address_space(1))) v4i_vs* as1_v4i;
typedef __attribute__((address_space(3))) v4i_vs* as3_v4i;
#endif

__device__ __forceinline__ void async_cp16(const unsigned short* g, unsigned short* l) {
#if HAVE_ASYNC_LDS
    __builtin_amdgcn_global_load_async_to_lds_b128((as1_v4i)g, (as3_v4i)l, 0, 0);
#else
    *(uint4*)l = *(const uint4*)g;
#endif
}
__device__ __forceinline__ void async_wait0() {
#if HAVE_ASYNC_LDS
#  if __has_builtin(__builtin_amdgcn_s_wait_asynccnt)
    __builtin_amdgcn_s_wait_asynccnt(0);
#  else
    asm volatile("s_wait_asynccnt 0x0" ::: "memory");
#  endif
#endif
}

__device__ __forceinline__ unsigned short f2bf(float f) {
    union { float f; unsigned u; } v; v.f = f;
    unsigned r = v.u + 0x7FFFu + ((v.u >> 16) & 1u);   // round-to-nearest-even
    return (unsigned short)(r >> 16);
}

__device__ __forceinline__ float gelu_tanh(float x) {
    const float c = 0.7978845608028654f;               // sqrt(2/pi)
    float t = tanhf(c * (x + 0.044715f * x * x * x));
    return 0.5f * x * (1.0f + t);
}

__device__ __forceinline__ f32x8 wmma_bf16(const FragBF& a, const FragBF& b, f32x8 c) {
    return __builtin_amdgcn_wmma_f32_16x16x32_bf16(false, a.v, false, b.v, (short)0, c, false, false);
}

// ---------------------------------------------------------------------------
// One-time weight prep: f32 [K x N] -> bf16 [N x K] (transpose + convert).
// Tiled 64x64 through LDS; coalesced on both sides.
// ---------------------------------------------------------------------------
__global__ void __launch_bounds__(256)
k_prep(const float* __restrict__ W, unsigned short* __restrict__ Wt, int K, int N) {
    __shared__ unsigned short tile[64][72];            // padded vs bank conflicts
    const float* Wm = W + (size_t)blockIdx.z * K * N;
    unsigned short* Wo = Wt + (size_t)blockIdx.z * K * N;
    int n0 = blockIdx.x * 64, k0 = blockIdx.y * 64;
    int tid = threadIdx.x;
    int nc = tid & 63, kr0 = tid >> 6;
    for (int kr = kr0; kr < 64; kr += 4)
        tile[kr][nc] = f2bf(Wm[(size_t)(k0 + kr) * N + n0 + nc]);
    __syncthreads();
    int nr = tid >> 2, kc0 = (tid & 3) << 4;
    union { unsigned short s[16]; uint4 q[2]; } o;
    #pragma unroll
    for (int j = 0; j < 16; ++j) o.s[j] = tile[kc0 + j][nr];
    *(uint4*)&Wo[(size_t)(n0 + nr) * K + k0 + kc0]     = o.q[0];
    *(uint4*)&Wo[(size_t)(n0 + nr) * K + k0 + kc0 + 8] = o.q[1];
}

// ---------------------------------------------------------------------------
__global__ void k_zero(float* __restrict__ racc, int* __restrict__ counts,
                       float* __restrict__ probsum, size_t n) {
    size_t i = (size_t)blockIdx.x * 256 + threadIdx.x;
    if (i < n) racc[i] = 0.0f;
    if (i < NE) { counts[i] = 0; probsum[i] = 0.0f; }
}

// ---------------------------------------------------------------------------
// Router: fused RMS norms, softmax over 8 logits, top-2, renormalized combine
// weights scattered into per-expert compacted lists + aux-loss statistics.
// ---------------------------------------------------------------------------
__global__ void __launch_bounds__(128)
k_router(const float* __restrict__ inputs, const float* __restrict__ orig,
         const float* __restrict__ pre_fwd, const float* __restrict__ pre_ln2,
         const float* __restrict__ rw,
         unsigned short* __restrict__ x_bf, unsigned short* __restrict__ routed_bf,
         int* __restrict__ counts, int* __restrict__ list,
         float* __restrict__ wlist, float* __restrict__ probsum) {
    __shared__ float red[9 * 128];
    __shared__ float s_rr;
    int t = blockIdx.x, tid = threadIdx.x;
    const float* xr = orig + (size_t)t * DD;
    const float* xi = inputs + (size_t)t * DD;

    float msq = 0.0f, lg[NE];
    #pragma unroll
    for (int e = 0; e < NE; ++e) lg[e] = 0.0f;
    for (int d = tid; d < DD; d += 128) {
        float v = xr[d];
        msq += v * v;
        float g = v * pre_fwd[d];
        const float* w = rw + (size_t)d * NE;
        #pragma unroll
        for (int e = 0; e < NE; ++e) lg[e] += g * w[e];
    }
    red[tid] = msq;
    #pragma unroll
    for (int e = 0; e < NE; ++e) red[(e + 1) * 128 + tid] = lg[e];
    __syncthreads();
    for (int s = 64; s > 0; s >>= 1) {
        if (tid < s)
            for (int v = 0; v < 9; ++v) red[v * 128 + tid] += red[v * 128 + tid + s];
        __syncthreads();
    }
    if (tid == 0) s_rr = rsqrtf(red[0] / (float)DD + EPSV);
    __syncthreads();
    float rr = s_rr;

    for (int d = tid; d < DD; d += 128) {
        routed_bf[(size_t)t * DD + d] = f2bf(xr[d] * rr * pre_ln2[d]);
        x_bf[(size_t)t * DD + d]      = f2bf(xi[d]);
    }

    if (tid == 0) {
        float scale = rr * rsqrtf((float)DD);
        float l[NE], mx = -1e30f;
        #pragma unroll
        for (int e = 0; e < NE; ++e) { l[e] = red[(e + 1) * 128] * scale; mx = fmaxf(mx, l[e]); }
        float se = 0.0f;
        #pragma unroll
        for (int e = 0; e < NE; ++e) { l[e] = expf(l[e] - mx); se += l[e]; }
        float inv = 1.0f / se;
        float p[NE]; int i0 = 0;
        #pragma unroll
        for (int e = 0; e < NE; ++e) {
            p[e] = l[e] * inv;
            atomicAdd(&probsum[e], p[e]);
            if (p[e] > p[i0]) i0 = e;
        }
        int i1 = (i0 == 0) ? 1 : 0;
        #pragma unroll
        for (int e = 0; e < NE; ++e) if (e != i0 && p[e] > p[i1]) i1 = e;
        float wsum = p[i0] + p[i1];
        int p0 = atomicAdd(&counts[i0], 1);
        list[i0 * TT + p0] = t;  wlist[i0 * TT + p0] = p[i0] / wsum;
        int p1 = atomicAdd(&counts[i1], 1);
        list[i1 * TT + p1] = t;  wlist[i1 * TT + p1] = p[i1] / wsum;
    }
}

// ---------------------------------------------------------------------------
// Dual GEMM: H = gelu(X @ W0) * (X @ W1).  X bf16 [rows x Dk] row-major,
// W*t bf16 [Nn x Dk] (pre-transposed).  Macro-tile 64(M) x 128(N), BK=32;
// 8 waves, each wave a 32x32 tile per weight (2 A-frags x 2 B-frags,
// 8 wmma / K-step).  segmented=1: rows gathered per expert via list/counts.
// ---------------------------------------------------------------------------
__global__ void __launch_bounds__(256)
k_gemm_h(const unsigned short* __restrict__ Xbf,
         const unsigned short* __restrict__ W0t, const unsigned short* __restrict__ W1t,
         unsigned short* __restrict__ H,
         const int* __restrict__ list, const int* __restrict__ counts,
         int Dk, int Nn, int segmented) {
    __shared__ __align__(16) unsigned short As[64 * 32];
    __shared__ __align__(16) unsigned short B0[128 * 32];
    __shared__ __align__(16) unsigned short B1[128 * 32];

    int tid = threadIdx.x;
    int ntile = blockIdx.x, rtile = blockIdx.y;
    int e = 0, seg = rtile, cnt = 0x7fffffff;
    size_t wbase = 0, hbase = 0;
    if (segmented) {
        e = rtile >> 6;                 // TT/64 = 64 row-tiles per expert
        seg = rtile & 63;
        cnt = counts[e];
        if (seg * 64 >= cnt) return;
        wbase = (size_t)e * Dk * Nn;
        hbase = (size_t)e * TT;
    }
    int nBase = ntile * 128;

    // A stage: 64 rows x 4 groups of 8 bf16 (one 16B copy / thread)
    int ar = tid >> 2, ac = (tid & 3) << 3;
    int r_local = seg * 64 + ar;
    bool avalid = true; int tok = r_local;
    if (segmented) { avalid = (r_local < cnt); tok = avalid ? list[e * TT + r_local] : 0; }
    const unsigned short* arow = Xbf + (size_t)tok * Dk + ac;
    unsigned short* aslot = &As[ar * 32 + ac];
    if (!avalid) { uint4 z = {0, 0, 0, 0}; *(uint4*)aslot = z; }  // stays zero

    // B stage: 128 n-rows x 2 groups of 16 bf16 (two 16B copies / thread / W)
    int bn = tid & 127, bk = (tid >> 7) << 4;
    const unsigned short* b0p = W0t + wbase + (size_t)(nBase + bn) * Dk + bk;
    const unsigned short* b1p = W1t + wbase + (size_t)(nBase + bn) * Dk + bk;
    unsigned short* b0s = &B0[bn * 32 + bk];
    unsigned short* b1s = &B1[bn * 32 + bk];

    // wave/fragment coords: 2 waves over M, 4 over N
    int wv = tid >> 5, ln = tid & 31;
    int wmq = wv >> 2, wnq = wv & 3;
    int abase = (ln >> 4) << 4;
    int arf0 = (wmq * 32 + (ln & 15)) * 32 + abase;
    int arf1 = (wmq * 32 + 16 + (ln & 15)) * 32 + abase;
    int kb = (ln >> 4) << 4;
    int brow0 = (wnq * 32 + (ln & 15)) * 32 + kb;
    int brow1 = (wnq * 32 + 16 + (ln & 15)) * 32 + kb;

    const f32x8 fz = {0, 0, 0, 0, 0, 0, 0, 0};
    f32x8 acc[8];                                   // [w*4 + mi*2 + ni]
    #pragma unroll
    for (int i = 0; i < 8; ++i) acc[i] = fz;

    for (int k0 = 0; k0 < Dk; k0 += 32) {
        if (avalid) async_cp16(arow + k0, aslot);
        async_cp16(b0p + k0,     b0s);
        async_cp16(b0p + k0 + 8, b0s + 8);
        async_cp16(b1p + k0,     b1s);
        async_cp16(b1p + k0 + 8, b1s + 8);
        async_wait0();
        __syncthreads();

        FragBF fa0, fa1, fb;
        fa0.q[0] = *(const uint4*)&As[arf0]; fa0.q[1] = *(const uint4*)&As[arf0 + 8];
        fa1.q[0] = *(const uint4*)&As[arf1]; fa1.q[1] = *(const uint4*)&As[arf1 + 8];
        fb.q[0] = *(const uint4*)&B0[brow0]; fb.q[1] = *(const uint4*)&B0[brow0 + 8];
        acc[0] = wmma_bf16(fa0, fb, acc[0]);
        acc[2] = wmma_bf16(fa1, fb, acc[2]);
        fb.q[0] = *(const uint4*)&B0[brow1]; fb.q[1] = *(const uint4*)&B0[brow1 + 8];
        acc[1] = wmma_bf16(fa0, fb, acc[1]);
        acc[3] = wmma_bf16(fa1, fb, acc[3]);
        fb.q[0] = *(const uint4*)&B1[brow0]; fb.q[1] = *(const uint4*)&B1[brow0 + 8];
        acc[4] = wmma_bf16(fa0, fb, acc[4]);
        acc[6] = wmma_bf16(fa1, fb, acc[6]);
        fb.q[0] = *(const uint4*)&B1[brow1]; fb.q[1] = *(const uint4*)&B1[brow1 + 8];
        acc[5] = wmma_bf16(fa0, fb, acc[5]);
        acc[7] = wmma_bf16(fa1, fb, acc[7]);
        __syncthreads();
    }

    #pragma unroll
    for (int mi = 0; mi < 2; ++mi) {
        int mrow0 = seg * 64 + wmq * 32 + mi * 16 + ((ln >> 4) << 3);
        #pragma unroll
        for (int ni = 0; ni < 2; ++ni) {
            int ncol = nBase + wnq * 32 + ni * 16 + (ln & 15);
            f32x8 g = acc[mi * 2 + ni];
            f32x8 u = acc[4 + mi * 2 + ni];
            #pragma unroll
            for (int i = 0; i < 8; ++i) {
                size_t row = hbase + (size_t)(mrow0 + i);
                H[row * Nn + ncol] = f2bf(gelu_tanh(g[i]) * u[i]);
            }
        }
    }
}

// ---------------------------------------------------------------------------
// Out GEMM: O = H @ Wo (Wot bf16 [Nn x Kk] pre-transposed).
// Plain: store f32.  Segmented: scatter-add w[row]*val via atomic f32.
// Macro-tile 64 x 128; each wave 32x32 (4 wmma / K-step).
// ---------------------------------------------------------------------------
__global__ void __launch_bounds__(256)
k_gemm_out(const unsigned short* __restrict__ H, const unsigned short* __restrict__ Wot,
           float* __restrict__ Oplain, float* __restrict__ Oacc,
           const int* __restrict__ list, const float* __restrict__ wlist,
           const int* __restrict__ counts, int Kk, int Nn, int segmented) {
    __shared__ __align__(16) unsigned short As[64 * 32];
    __shared__ __align__(16) unsigned short Bs[128 * 32];

    int tid = threadIdx.x;
    int ntile = blockIdx.x, rtile = blockIdx.y;
    int e = 0, seg = rtile, cnt = 0x7fffffff;
    size_t wbase = 0, hbase = 0;
    if (segmented) {
        e = rtile >> 6;
        seg = rtile & 63;
        cnt = counts[e];
        if (seg * 64 >= cnt) return;
        wbase = (size_t)e * Kk * Nn;
        hbase = (size_t)e * TT;
    }
    int nBase = ntile * 128;

    int ar = tid >> 2, ac = (tid & 3) << 3;
    const unsigned short* arow = H + (hbase + (size_t)(seg * 64 + ar)) * Kk + ac;
    unsigned short* aslot = &As[ar * 32 + ac];

    int bn = tid & 127, bk = (tid >> 7) << 4;
    const unsigned short* bp = Wot + wbase + (size_t)(nBase + bn) * Kk + bk;
    unsigned short* bs = &Bs[bn * 32 + bk];

    int wv = tid >> 5, ln = tid & 31;
    int wmq = wv >> 2, wnq = wv & 3;
    int abase = (ln >> 4) << 4;
    int arf0 = (wmq * 32 + (ln & 15)) * 32 + abase;
    int arf1 = (wmq * 32 + 16 + (ln & 15)) * 32 + abase;
    int kb = (ln >> 4) << 4;
    int brow0 = (wnq * 32 + (ln & 15)) * 32 + kb;
    int brow1 = (wnq * 32 + 16 + (ln & 15)) * 32 + kb;

    const f32x8 fz = {0, 0, 0, 0, 0, 0, 0, 0};
    f32x8 acc[4];                                   // [mi*2 + ni]
    #pragma unroll
    for (int i = 0; i < 4; ++i) acc[i] = fz;

    for (int k0 = 0; k0 < Kk; k0 += 32) {
        async_cp16(arow + k0, aslot);               // padded rows hold zeros
        async_cp16(bp + k0,     bs);
        async_cp16(bp + k0 + 8, bs + 8);
        async_wait0();
        __syncthreads();

        FragBF fa0, fa1, fb;
        fa0.q[0] = *(const uint4*)&As[arf0]; fa0.q[1] = *(const uint4*)&As[arf0 + 8];
        fa1.q[0] = *(const uint4*)&As[arf1]; fa1.q[1] = *(const uint4*)&As[arf1 + 8];
        fb.q[0] = *(const uint4*)&Bs[brow0]; fb.q[1] = *(const uint4*)&Bs[brow0 + 8];
        acc[0] = wmma_bf16(fa0, fb, acc[0]);
        acc[2] = wmma_bf16(fa1, fb, acc[2]);
        fb.q[0] = *(const uint4*)&Bs[brow1]; fb.q[1] = *(const uint4*)&Bs[brow1 + 8];
        acc[1] = wmma_bf16(fa0, fb, acc[1]);
        acc[3] = wmma_bf16(fa1, fb, acc[3]);
        __syncthreads();
    }

    #pragma unroll
    for (int mi = 0; mi < 2; ++mi) {
        int mrow0 = seg * 64 + wmq * 32 + mi * 16 + ((ln >> 4) << 3);
        #pragma unroll
        for (int ni = 0; ni < 2; ++ni) {
            int ncol = nBase + wnq * 32 + ni * 16 + (ln & 15);
            f32x8 a = acc[mi * 2 + ni];
            if (!segmented) {
                #pragma unroll
                for (int i = 0; i < 8; ++i)
                    Oplain[(size_t)(mrow0 + i) * Nn + ncol] = a[i];
            } else {
                #pragma unroll
                for (int i = 0; i < 8; ++i) {
                    int r = mrow0 + i;
                    if (r < cnt) {
                        int tokn = list[e * TT + r];
                        float w  = wlist[e * TT + r];
                        atomicAdd(&Oacc[(size_t)tokn * Nn + ncol], w * a[i]);
                    }
                }
            }
        }
    }
}

// ---------------------------------------------------------------------------
__global__ void __launch_bounds__(256)
k_epilogue(const float* __restrict__ shf, const float* __restrict__ racc,
           const float* __restrict__ post1, const float* __restrict__ post2,
           float* __restrict__ out) {
    __shared__ float r1[256], r2[256];
    int t = blockIdx.x, tid = threadIdx.x;
    const float* sr = shf + (size_t)t * DD;
    const float* rr = racc + (size_t)t * DD;
    float a = 0.0f, b = 0.0f;
    for (int d = tid; d < DD; d += 256) {
        float s = sr[d]; a += s * s;
        float r = rr[d]; b += r * r;
    }
    r1[tid] = a; r2[tid] = b;
    __syncthreads();
    for (int s = 128; s > 0; s >>= 1) {
        if (tid < s) { r1[tid] += r1[tid + s]; r2[tid] += r2[tid + s]; }
        __syncthreads();
    }
    float rs1 = rsqrtf(r1[0] / (float)DD + EPSV);
    float rs2 = rsqrtf(r2[0] / (float)DD + EPSV);
    for (int d = tid; d < DD; d += 256)
        out[(size_t)t * DD + d] = rr[d] * rs2 * post2[d] + sr[d] * rs1 * post1[d];
}

__global__ void k_lb(const int* __restrict__ counts, const float* __restrict__ probsum,
                     float* __restrict__ out) {
    if (threadIdx.x == 0 && blockIdx.x == 0) {
        float lb = 0.0f;
        for (int e = 0; e < NE; ++e)
            lb += ((float)counts[e] / (float)(TT * TOPK)) * (probsum[e] / (float)TT);
        out[(size_t)TT * DD] = lb * (float)NE;
    }
}

// ---------------------------------------------------------------------------
extern "C" void kernel_launch(void* const* d_in, const int* in_sizes, int n_in,
                              void* d_out, int out_size, void* d_ws, size_t ws_size,
                              hipStream_t stream) {
    const float* inputs  = (const float*)d_in[0];
    const float* orig    = (const float*)d_in[1];
    const float* pre_fwd = (const float*)d_in[2];
    const float* pre_ln2 = (const float*)d_in[3];
    const float* post1   = (const float*)d_in[4];
    const float* post2   = (const float*)d_in[5];
    const float* rw      = (const float*)d_in[6];
    const float* e_wi0   = (const float*)d_in[7];
    const float* e_wi1   = (const float*)d_in[8];
    const float* e_wo    = (const float*)d_in[9];
    const float* s_wi0   = (const float*)d_in[10];
    const float* s_wi1   = (const float*)d_in[11];
    const float* s_wo    = (const float*)d_in[12];
    float* out = (float*)d_out;

    char* ws = (char*)d_ws;
    size_t off = 0;
    auto alloc = [&](size_t bytes) -> char* {
        char* p = ws + off;
        off = (off + bytes + 255) & ~(size_t)255;
        return p;
    };
    unsigned short* x_bf      = (unsigned short*)alloc((size_t)TT * DD * 2);
    unsigned short* routed_bf = (unsigned short*)alloc((size_t)TT * DD * 2);
    unsigned short* h_sh      = (unsigned short*)alloc((size_t)TT * FF * 2);
    float*          shared_f  = (float*)alloc((size_t)TT * DD * 4);
    unsigned short* h_moe     = (unsigned short*)alloc((size_t)NE * TT * MM * 2);
    float*          racc      = (float*)alloc((size_t)TT * DD * 4);
    int*            list      = (int*)alloc((size_t)NE * TT * 4);
    float*          wlist     = (float*)alloc((size_t)NE * TT * 4);
    int*            counts    = (int*)alloc(64);
    float*          probsum   = (float*)alloc(64);
    // pre-transposed bf16 weights [N x K]
    unsigned short* swi0_t = (unsigned short*)alloc((size_t)DD * FF * 2);
    unsigned short* swi1_t = (unsigned short*)alloc((size_t)DD * FF * 2);
    unsigned short* swo_t  = (unsigned short*)alloc((size_t)FF * DD * 2);
    unsigned short* ewi0_t = (unsigned short*)alloc((size_t)NE * DD * MM * 2);
    unsigned short* ewi1_t = (unsigned short*)alloc((size_t)NE * DD * MM * 2);
    unsigned short* ewo_t  = (unsigned short*)alloc((size_t)NE * MM * DD * 2);

    // weight prep (transpose + f32->bf16)
    k_prep<<<dim3(FF / 64, DD / 64, 1),  256, 0, stream>>>(s_wi0, swi0_t, DD, FF);
    k_prep<<<dim3(FF / 64, DD / 64, 1),  256, 0, stream>>>(s_wi1, swi1_t, DD, FF);
    k_prep<<<dim3(DD / 64, FF / 64, 1),  256, 0, stream>>>(s_wo,  swo_t,  FF, DD);
    k_prep<<<dim3(MM / 64, DD / 64, NE), 256, 0, stream>>>(e_wi0, ewi0_t, DD, MM);
    k_prep<<<dim3(MM / 64, DD / 64, NE), 256, 0, stream>>>(e_wi1, ewi1_t, DD, MM);
    k_prep<<<dim3(DD / 64, MM / 64, NE), 256, 0, stream>>>(e_wo,  ewo_t,  MM, DD);

    k_zero<<<(TT * DD + 255) / 256, 256, 0, stream>>>(racc, counts, probsum, (size_t)TT * DD);
    k_router<<<TT, 128, 0, stream>>>(inputs, orig, pre_fwd, pre_ln2, rw,
                                     x_bf, routed_bf, counts, list, wlist, probsum);
    // shared expert
    k_gemm_h<<<dim3(FF / 128, TT / 64), 256, 0, stream>>>(
        x_bf, swi0_t, swi1_t, h_sh, (const int*)nullptr, (const int*)nullptr, DD, FF, 0);
    k_gemm_out<<<dim3(DD / 128, TT / 64), 256, 0, stream>>>(
        h_sh, swo_t, shared_f, (float*)nullptr, (const int*)nullptr,
        (const float*)nullptr, (const int*)nullptr, FF, DD, 0);
    // routed experts (gathered top-2 tokens per expert)
    k_gemm_h<<<dim3(MM / 128, NE * TT / 64), 256, 0, stream>>>(
        routed_bf, ewi0_t, ewi1_t, h_moe, list, counts, DD, MM, 1);
    k_gemm_out<<<dim3(DD / 128, NE * TT / 64), 256, 0, stream>>>(
        h_moe, ewo_t, (float*)nullptr, racc, list, wlist, counts, MM, DD, 1);
    k_epilogue<<<TT, 256, 0, stream>>>(shared_f, racc, post1, post2, out);
    k_lb<<<1, 32, 0, stream>>>(counts, probsum, out);
}
